// GraphClassifier_1949915152972
// MI455X (gfx1250) — compile-verified, matrix operand order checked by reading, not run on hardware
//
#include <hip/hip_runtime.h>
#include <hip/hip_bf16.h>

typedef __attribute__((ext_vector_type(16))) _Float16 v16h;
typedef __attribute__((ext_vector_type(8)))  _Float16 v8h;
typedef __attribute__((ext_vector_type(8)))  float    v8f;

#define N_NODES 50000
#define N_EDGES 800000
#define IN_C    64
#define HID     128
#define OUT_C   2
#define N_GRAPHS 64

// ---------------- utility kernels ----------------

__global__ void k_zero_f32(float* p, int n) {
    int i = blockIdx.x * blockDim.x + threadIdx.x;
    if (i < n) p[i] = 0.0f;
}

__global__ void k_fill_f32(float* p, float v, int n) {
    int i = blockIdx.x * blockDim.x + threadIdx.x;
    if (i < n) p[i] = v;
}

// deg[dst] += 1 for every real edge (self-loop contribution pre-filled as 1.0)
__global__ void k_deg(const int* dst, float* deg, int E) {
    int i = blockIdx.x * blockDim.x + threadIdx.x;
    if (i < E) atomicAdd(&deg[dst[i]], 1.0f);
}

// in-place deg -> 1/sqrt(deg)   (deg >= 1 always due to self loop)
__global__ void k_rsqrt(float* d, int n) {
    int i = blockIdx.x * blockDim.x + threadIdx.x;
    if (i < n) d[i] = rsqrtf(d[i]);
}

// per-graph node counts
__global__ void k_count(const int* batch, float* cnt, int n) {
    int i = blockIdx.x * blockDim.x + threadIdx.x;
    if (i < n) atomicAdd(&cnt[batch[i]], 1.0f);
}

// fp32 -> fp16 convert
__global__ void k_cvt_h(const float* x, _Float16* y, int n) {
    int i = blockIdx.x * blockDim.x + threadIdx.x;
    if (i < n) y[i] = (_Float16)x[i];
}

// ---------------- weight fragment packing ----------------
// Pack W[K x Nc] (fp32, row-major) into per-(coltile, kstep, lane) v16h
// fragments matching the CDNA5 16-bit B-matrix (32x16) WMMA layout:
//   lanes 0-15  : col n = lane,    K-elems {kk+0..7, kk+16..23}
//   lanes 16-31 : col n = lane-16, K-elems {kk+8..15, kk+24..31}
__global__ void k_pack_w(const float* W, v16h* Wp, int ksteps, int Nc) {
    int idx  = blockIdx.x * blockDim.x + threadIdx.x;
    int lane = idx & 31;
    int t    = idx >> 5;
    int ks   = t % ksteps;
    int nt   = t / ksteps;
    if (nt >= Nc / 16) return;
    int n  = nt * 16 + (lane & 15);
    int kb = (lane >> 4) * 8;
    int k0 = ks * 32 + kb;
    v16h o;
#pragma unroll
    for (int j = 0; j < 8; ++j) {
        o[j]     = (_Float16)W[(size_t)(k0 + j)      * Nc + n];
        o[8 + j] = (_Float16)W[(size_t)(k0 + 16 + j) * Nc + n];
    }
    Wp[idx] = o;
}

// ---------------- WMMA GEMM: C[N x 128] = A[N x K](f16) @ Wp(f16 packed) ----
// One wave computes one 16x16 tile; 8 waves/block = all 128 output cols.
// gridDim.x = N/16 row tiles.
__global__ void __launch_bounds__(256)
k_gemm_wmma(const _Float16* __restrict__ A, const v16h* __restrict__ Wp,
            float* __restrict__ C, int K, int ksteps) {
    const int lane = threadIdx.x & 31;
    const int wave = threadIdx.x >> 5;      // col tile 0..7
    const int m0   = blockIdx.x * 16;
    const int n0   = wave * 16;
    const int half = lane >> 4;             // 0: lanes 0-15, 1: lanes 16-31
    const int mrow = m0 + (lane & 15);
    const int kb   = half * 8;

    v8f acc = {};
    for (int ks = 0; ks < ksteps; ++ks) {
        const int kk = ks * 32;
        const _Float16* ap = A + (size_t)mrow * K + kk + kb;
        v8h lo = *(const v8h*)(ap);
        v8h hi = *(const v8h*)(ap + 16);
        v16h a = __builtin_shufflevector(lo, hi, 0,1,2,3,4,5,6,7,8,9,10,11,12,13,14,15);
        v16h b = Wp[(size_t)(wave * ksteps + ks) * 32 + lane];
        acc = __builtin_amdgcn_wmma_f32_16x16x32_f16(
            /*neg_a=*/false, a, /*neg_b=*/false, b,
            /*c_mod=*/(short)0, acc, /*reuse_a=*/false, /*reuse_b=*/false);
    }
#pragma unroll
    for (int r = 0; r < 8; ++r) {
        C[(size_t)(m0 + r + half * 8) * HID + n0 + (lane & 15)] = acc[r];
    }
}

// ---------------- edge gather + normalized scatter-add ----------------
// one wave (32 lanes) per edge, float4 per lane over 128 channels;
// agg[dst] += H[src] * dinv[src]*dinv[dst].   L2-resident atomics.
__global__ void k_edge_scatter(const float* __restrict__ H,
                               const int* __restrict__ src,
                               const int* __restrict__ dst,
                               const float* __restrict__ dinv,
                               float* __restrict__ agg, int E) {
    int tid  = blockIdx.x * blockDim.x + threadIdx.x;
    int e    = tid >> 5;
    int lane = tid & 31;
    if (e >= E) return;
    int s = src[e], d = dst[e];
    float nrm = dinv[s] * dinv[d];
    const float4* hp = (const float4*)(H + (size_t)s * HID);
    float4 v = hp[lane];
    float* op = agg + (size_t)d * HID + lane * 4;
    atomicAdd(op + 0, v.x * nrm);
    atomicAdd(op + 1, v.y * nrm);
    atomicAdd(op + 2, v.z * nrm);
    atomicAdd(op + 3, v.w * nrm);
}

// conv1 epilogue: + self-loop + bias, ReLU, convert to fp16 for next GEMM
__global__ void k_finish1(const float* __restrict__ H, const float* __restrict__ agg,
                          const float* __restrict__ dinv, const float* __restrict__ b,
                          _Float16* __restrict__ outh, int n) {
    int i = blockIdx.x * blockDim.x + threadIdx.x;
    if (i >= n) return;
    int node = i >> 7, c = i & (HID - 1);
    float di = dinv[node];
    float v = agg[i] + H[i] * di * di + b[c];
    outh[i] = (_Float16)fmaxf(v, 0.0f);
}

// conv2 epilogue: + self-loop + bias (no ReLU), atomic-add into per-graph pool
__global__ void k_finish2(const float* __restrict__ H, const float* __restrict__ agg,
                          const float* __restrict__ dinv, const float* __restrict__ b,
                          const int* __restrict__ batch, float* __restrict__ pool, int n) {
    int i = blockIdx.x * blockDim.x + threadIdx.x;
    if (i >= n) return;
    int node = i >> 7, c = i & (HID - 1);
    float di = dinv[node];
    float v = agg[i] + H[i] * di * di + b[c];
    atomicAdd(&pool[(size_t)batch[node] * HID + c], v);
}

__global__ void k_pool_div(const float* sums, const float* cnt, float* g, int n) {
    int i = blockIdx.x * blockDim.x + threadIdx.x;
    if (i >= n) return;
    g[i] = sums[i] / fmaxf(cnt[i >> 7], 1.0f);
}

// MLP head (tiny: 64x128 @ 128x128 then 64x128 @ 128x2)
__global__ void k_head1(const float* g, const float* Wm1, const float* bm1,
                        float* gh, int n) {
    int i = blockIdx.x * blockDim.x + threadIdx.x;
    if (i >= n) return;
    int gi = i >> 7, j = i & (HID - 1);
    float acc = bm1[j];
    for (int k = 0; k < HID; ++k) acc += g[(size_t)gi * HID + k] * Wm1[(size_t)k * HID + j];
    gh[i] = fmaxf(acc, 0.0f);
}

__global__ void k_head2(const float* gh, const float* Wm2, const float* bm2,
                        float* out, int n) {
    int i = blockIdx.x * blockDim.x + threadIdx.x;
    if (i >= n) return;
    int gi = i / OUT_C, j = i % OUT_C;
    float acc = bm2[j];
    for (int k = 0; k < HID; ++k) acc += gh[(size_t)gi * HID + k] * Wm2[(size_t)k * OUT_C + j];
    out[i] = acc;
}

// ---------------- launcher ----------------

static inline size_t alignup(size_t x) { return (x + 255) & ~(size_t)255; }

extern "C" void kernel_launch(void* const* d_in, const int* in_sizes, int n_in,
                              void* d_out, int out_size, void* d_ws, size_t ws_size,
                              hipStream_t stream) {
    (void)in_sizes; (void)n_in; (void)out_size; (void)ws_size;
    const float* x    = (const float*)d_in[0];
    const int*   ei   = (const int*)d_in[1];     // [2, E]
    const int*   batch= (const int*)d_in[2];
    const float* W1   = (const float*)d_in[3];
    const float* b1   = (const float*)d_in[4];
    const float* W2   = (const float*)d_in[5];
    const float* b2   = (const float*)d_in[6];
    const float* Wm1  = (const float*)d_in[7];
    const float* bm1  = (const float*)d_in[8];
    const float* Wm2  = (const float*)d_in[9];
    const float* bm2  = (const float*)d_in[10];
    const int* esrc = ei;
    const int* edst = ei + N_EDGES;
    float* out = (float*)d_out;

    // bump-allocate workspace
    char* ws = (char*)d_ws;
    size_t off = 0;
    auto alloc = [&](size_t bytes) { char* p = ws + off; off += alignup(bytes); return p; };

    float*    dinv = (float*)   alloc((size_t)N_NODES * 4);                 // deg -> dinv
    _Float16* xh   = (_Float16*)alloc((size_t)N_NODES * IN_C * 2);
    float*    H    = (float*)   alloc((size_t)N_NODES * HID * 4);           // GEMM out (reused)
    float*    agg  = (float*)   alloc((size_t)N_NODES * HID * 4);           // scatter accumulator
    _Float16* h1h  = (_Float16*)alloc((size_t)N_NODES * HID * 2);
    v16h*     W1p  = (v16h*)    alloc((size_t)(HID/16) * (IN_C/32) * 32 * sizeof(v16h));
    v16h*     W2p  = (v16h*)    alloc((size_t)(HID/16) * (HID/32) * 32 * sizeof(v16h));
    float*    pool = (float*)   alloc((size_t)N_GRAPHS * HID * 4);
    float*    cnt  = (float*)   alloc((size_t)N_GRAPHS * 4);
    float*    g    = (float*)   alloc((size_t)N_GRAPHS * HID * 4);
    float*    gh   = (float*)   alloc((size_t)N_GRAPHS * HID * 4);

    const int B = 256;
    const int NH = N_NODES * HID;

    // --- normalization factors ---
    k_fill_f32<<<(N_NODES + B - 1) / B, B, 0, stream>>>(dinv, 1.0f, N_NODES);
    k_deg<<<(N_EDGES + B - 1) / B, B, 0, stream>>>(edst, dinv, N_EDGES);
    k_rsqrt<<<(N_NODES + B - 1) / B, B, 0, stream>>>(dinv, N_NODES);

    // --- pool counts ---
    k_zero_f32<<<(N_GRAPHS * HID + N_GRAPHS + B - 1) / B, B, 0, stream>>>(pool, N_GRAPHS * HID + N_GRAPHS); // pool+cnt contiguous? not guaranteed
    k_zero_f32<<<1, N_GRAPHS, 0, stream>>>(cnt, N_GRAPHS);
    k_zero_f32<<<(N_GRAPHS * HID + B - 1) / B, B, 0, stream>>>(pool, N_GRAPHS * HID);
    k_count<<<(N_NODES + B - 1) / B, B, 0, stream>>>(batch, cnt, N_NODES);

    // --- conv1: H = relu(scatter(norm * (X@W1)[src]) + self + b1) ---
    k_cvt_h<<<(N_NODES * IN_C + B - 1) / B, B, 0, stream>>>(x, xh, N_NODES * IN_C);
    k_pack_w<<<((HID/16) * (IN_C/32) * 32 + B - 1) / B, B, 0, stream>>>(W1, W1p, IN_C / 32, HID);
    k_gemm_wmma<<<N_NODES / 16, 256, 0, stream>>>(xh, W1p, H, IN_C, IN_C / 32);
    k_zero_f32<<<(NH + B - 1) / B, B, 0, stream>>>(agg, NH);
    k_edge_scatter<<<(N_EDGES * 32 + B - 1) / B, B, 0, stream>>>(H, esrc, edst, dinv, agg, N_EDGES);
    k_finish1<<<(NH + B - 1) / B, B, 0, stream>>>(H, agg, dinv, b1, h1h, NH);

    // --- conv2: pool += scatter(norm * (h1@W2)[src]) + self + b2 ---
    k_pack_w<<<((HID/16) * (HID/32) * 32 + B - 1) / B, B, 0, stream>>>(W2, W2p, HID / 32, HID);
    k_gemm_wmma<<<N_NODES / 16, 256, 0, stream>>>(h1h, W2p, H, HID, HID / 32);
    k_zero_f32<<<(NH + B - 1) / B, B, 0, stream>>>(agg, NH);
    k_edge_scatter<<<(N_EDGES * 32 + B - 1) / B, B, 0, stream>>>(H, esrc, edst, dinv, agg, N_EDGES);
    k_finish2<<<(NH + B - 1) / B, B, 0, stream>>>(H, agg, dinv, b2, batch, pool, NH);

    // --- mean pool + head ---
    k_pool_div<<<(N_GRAPHS * HID + B - 1) / B, B, 0, stream>>>(pool, cnt, g, N_GRAPHS * HID);
    k_head1<<<(N_GRAPHS * HID + B - 1) / B, B, 0, stream>>>(g, Wm1, bm1, gh, N_GRAPHS * HID);
    k_head2<<<(N_GRAPHS * OUT_C + B - 1) / B, B, 0, stream>>>(gh, Wm2, bm2, out, N_GRAPHS * OUT_C);
}
